// SpeechResModel_28621662061096
// MI455X (gfx1250) — matrix operand order repaired and checked by reading, not compiled, not run on hardware
//
#include <hip/hip_runtime.h>
#include <hip/hip_bf16.h>

typedef __attribute__((ext_vector_type(16))) _Float16 v16h;
typedef __attribute__((ext_vector_type(8)))  _Float16 v8h;
typedef __attribute__((ext_vector_type(8)))  float    v8f;

#define TT 4096
#define BB 64
#define FF 40
#define HH 128
#define NCC 12

union U16 { v16h v; struct { v8h lo; v8h hi; } s; _Float16 h[16]; };

__device__ __forceinline__ v8f wmma16(v16h a, v16h b, v8f c) {
  return __builtin_amdgcn_wmma_f32_16x16x32_f16(false, a, false, b, (short)0, c, false, false);
}

// Hardware tanh (V_TANH_F32) when available; sigmoid via tanh identity.
#if defined(__has_builtin)
#if __has_builtin(__builtin_amdgcn_tanhf)
#define FAST_TANH(x) __builtin_amdgcn_tanhf(x)
#endif
#endif
#ifndef FAST_TANH
#define FAST_TANH(x) tanhf(x)
#endif
__device__ __forceinline__ float fast_tanh(float x) { return FAST_TANH(x); }
__device__ __forceinline__ float sigm(float x) { return 0.5f * FAST_TANH(0.5f * x) + 0.5f; }

// A fragment (16x32 f16): lane holds row = rbase + lane%16;
// halves[0..7]  = src[row][k0 + kA + 0..7],  halves[8..15] = src[row][k0 + kA + 16..23], kA = (lane>=16)?8:0
__device__ __forceinline__ v16h load_afrag_g(const _Float16* __restrict__ base, int strideH,
                                             int row, int k0, int lane) {
  int kA = (lane >= 16) ? 8 : 0;
  const _Float16* p = base + (size_t)row * strideH + k0 + kA;
  U16 u; u.s.lo = *(const v8h*)(p); u.s.hi = *(const v8h*)(p + 16);
  return u.v;
}

// B fragment (32x16 f16): lane holds col n = nbase + lane%16;
// halves[0..7] = W[n][k0 + kB + 0..7], halves[8..15] = W[n][k0 + kB + 16..23], kB = (lane>=16)?8:0   (W stored n-major: B[k][n] = W[n][k])
__device__ __forceinline__ v16h load_bfrag_g(const _Float16* __restrict__ base, int strideH,
                                             int ncol, int k0, int lane) {
  int kB = (lane >= 16) ? 8 : 0;
  const _Float16* p = base + (size_t)ncol * strideH + k0 + kB;
  U16 u; u.s.lo = *(const v8h*)(p); u.s.hi = *(const v8h*)(p + 16);
  return u.v;
}

// ---------------- weight conversion / packing ----------------
__global__ void convert_kernel(const float* __restrict__ Wih0, const float* __restrict__ Wih1,
                               const float* __restrict__ Whh0, const float* __restrict__ Whh1,
                               const float* __restrict__ weightW,
                               const float* __restrict__ bih0, const float* __restrict__ bhh0,
                               const float* __restrict__ bih1, const float* __restrict__ bhh1,
                               _Float16* __restrict__ wih0p, _Float16* __restrict__ wih1c,
                               _Float16* __restrict__ whh0c, _Float16* __restrict__ whh1c,
                               _Float16* __restrict__ wWt,
                               float* __restrict__ bias0c, float* __restrict__ bias1c) {
  int tid = blockIdx.x * blockDim.x + threadIdx.x;       // 65536 threads
  if (tid < 512 * 64) {                                  // W_ih0 [512,40] -> padded [512,64]
    int r = tid >> 6, k = tid & 63;
    wih0p[tid] = (k < FF) ? (_Float16)Wih0[r * FF + k] : (_Float16)0.f;
  }
  if (tid < 512 * 128) {
    wih1c[tid] = (_Float16)Wih1[tid];
    whh0c[tid] = (_Float16)Whh0[tid];
    whh1c[tid] = (_Float16)Whh1[tid];
  }
  if (tid < 128 * 128) {                                 // wWt[n][k] = weight_W[k][n]
    int n = tid >> 7, k = tid & 127;
    wWt[tid] = (_Float16)weightW[k * 128 + n];
  }
  if (tid < 512) {
    bias0c[tid] = bih0[tid] + bhh0[tid];
    bias1c[tid] = bih1[tid] + bhh1[tid];
  }
}

// ---------------- input projection, layer 0: xpf = pack(x @ W_ih0^T + b) ----------------
// xpf fragment layout (v8h units): [((t*32 + w)*4 + g)*32 + lane]
__global__ __launch_bounds__(256) void proj0_kernel(const float* __restrict__ x,      // [B][T][F]
                                                    const _Float16* __restrict__ wih, // [512][64]
                                                    const float* __restrict__ biasc,  // [512]
                                                    _Float16* __restrict__ xpf) {
  int gtid = blockIdx.x * blockDim.x + threadIdx.x;
  int lane = gtid & 31;
  int gw = gtid >> 5;
  int w = gw & 31, t = gw >> 5;
  if (t >= TT) return;
  int mtile = w & 3, cg = w >> 2;
  int c0 = cg * 16, l15 = lane & 15;
  int kA = (lane >= 16) ? 8 : 0;
  int b = mtile * 16 + l15;
  const float* xrow = x + ((size_t)b * TT + t) * FF;
  v16h Af[2];
#pragma unroll
  for (int kk = 0; kk < 2; ++kk) {
    U16 u;
#pragma unroll
    for (int j = 0; j < 8; ++j) {
      int k0 = kk * 32 + kA + j;
      int k1 = k0 + 16;
      u.h[j]     = (k0 < FF) ? (_Float16)xrow[k0] : (_Float16)0.f;
      u.h[8 + j] = (k1 < FF) ? (_Float16)xrow[k1] : (_Float16)0.f;
    }
    Af[kk] = u.v;
  }
  v8h* xp = (v8h*)xpf + (((size_t)t * 32 + w) * 4) * 32;
#pragma unroll
  for (int g = 0; g < 4; ++g) {
    int n = g * 128 + c0 + l15;
    v8f acc = {0.f, 0.f, 0.f, 0.f, 0.f, 0.f, 0.f, 0.f};
#pragma unroll
    for (int kk = 0; kk < 2; ++kk)
      acc = wmma16(Af[kk], load_bfrag_g(wih, 64, n, kk * 32, lane), acc);
    float bn = biasc[n];
    v8h out;
#pragma unroll
    for (int v = 0; v < 8; ++v) out[v] = (_Float16)(acc[v] + bn);
    xp[g * 32 + lane] = out;
  }
}

// ---------------- input projection, layer 1: xpf = pack(h0 @ W_ih1^T + b) ----------------
__global__ __launch_bounds__(256) void proj1_kernel(const _Float16* __restrict__ hin, // [T*B][128]
                                                    const _Float16* __restrict__ wih, // [512][128]
                                                    const float* __restrict__ biasc,
                                                    _Float16* __restrict__ xpf) {
  int gtid = blockIdx.x * blockDim.x + threadIdx.x;
  int lane = gtid & 31;
  int gw = gtid >> 5;
  int w = gw & 31, t = gw >> 5;
  if (t >= TT) return;
  int mtile = w & 3, cg = w >> 2;
  int c0 = cg * 16, l15 = lane & 15;
  int r = t * 64 + mtile * 16 + l15;
  v16h Af[4];
#pragma unroll
  for (int kk = 0; kk < 4; ++kk) Af[kk] = load_afrag_g(hin, 128, r, kk * 32, lane);
  v8h* xp = (v8h*)xpf + (((size_t)t * 32 + w) * 4) * 32;
#pragma unroll
  for (int g = 0; g < 4; ++g) {
    int n = g * 128 + c0 + l15;
    v8f acc = {0.f, 0.f, 0.f, 0.f, 0.f, 0.f, 0.f, 0.f};
#pragma unroll
    for (int kk = 0; kk < 4; ++kk)
      acc = wmma16(Af[kk], load_bfrag_g(wih, 128, n, kk * 32, lane), acc);
    float bn = biasc[n];
    v8h out;
#pragma unroll
    for (int v = 0; v < 8; ++v) out[v] = (_Float16)(acc[v] + bn);
    xp[g * 32 + lane] = out;
  }
}

// ---------------- persistent LSTM recurrence: one workgroup, 32 waves, 4096 steps ----------------
// wave w: mtile = w&3 (rows m0..m0+15 of batch), cg = w>>2 (cols c0..c0+15 of H),
// computes ALL FOUR gate tiles (n = g*128 + c0) so gate math is in-register; c stays in VGPRs.
// Input-projection fragments are register double-buffered (t+1 loads issued during step t) and
// software-prefetched 8 steps ahead.
__global__ __launch_bounds__(1024) void lstm_rec_kernel(const _Float16* __restrict__ whh, // [512][128]
                                                        const _Float16* __restrict__ xpf,
                                                        _Float16* __restrict__ hout) {   // [T*B][128]
  __shared__ __align__(32) _Float16 hsh[64][136];   // padded stride vs bank conflicts
  int tid = threadIdx.x;
  int lane = tid & 31;
  int w = tid >> 5;
  int mtile = w & 3, cg = w >> 2;
  int m0 = mtile * 16, c0 = cg * 16;
  int l15 = lane & 15;
  int hi8 = (lane >= 16) ? 8 : 0;
  int kA = (lane >= 16) ? 8 : 0;
  int arow = m0 + l15;

  for (int i = tid; i < 64 * 136; i += 1024) (&hsh[0][0])[i] = (_Float16)0.f;

  // W_hh fragments: register-resident for the whole sequence (4 gates x 4 k-steps)
  v16h Bf[4][4];
#pragma unroll
  for (int g = 0; g < 4; ++g) {
    int n = g * 128 + c0 + l15;
#pragma unroll
    for (int kk = 0; kk < 4; ++kk) Bf[g][kk] = load_bfrag_g(whh, 128, n, kk * 32, lane);
  }
  float cst[8];
#pragma unroll
  for (int v = 0; v < 8; ++v) cst[v] = 0.f;

  const v8h* xpbase = (const v8h*)xpf + (size_t)w * 4 * 32 + lane;   // + t*4096 + g*32
  // preload step 0 projection fragments
  v8h xcur[4];
#pragma unroll
  for (int g = 0; g < 4; ++g) xcur[g] = xpbase[g * 32];

  __syncthreads();

  for (int t = 0; t < TT; ++t) {
    // prefetch projection stream ~8 steps ahead (one line per gate across the wave)
    if (t + 8 < TT) {
      const v8h* xpp = xpbase + (size_t)(t + 8) * 4096;
#pragma unroll
      for (int g = 0; g < 4; ++g) __builtin_prefetch((const void*)(xpp + g * 32), 0, 0);
    }
    // issue t+1 fragment loads now; waited on only at end of this step
    v8h xnext[4];
    if (t + 1 < TT) {
      const v8h* xpn = xpbase + (size_t)(t + 1) * 4096;
#pragma unroll
      for (int g = 0; g < 4; ++g) xnext[g] = xpn[g * 32];
    }

    // A fragments from LDS h (prev step)
    v16h Af[4];
#pragma unroll
    for (int kk = 0; kk < 4; ++kk) {
      const _Float16* p = &hsh[arow][kk * 32 + kA];
      U16 u; u.s.lo = *(const v8h*)(p); u.s.hi = *(const v8h*)(p + 16);
      Af[kk] = u.v;
    }
    // seed C with current (preloaded) input-projection fragments
    v8f acc[4];
#pragma unroll
    for (int g = 0; g < 4; ++g) {
#pragma unroll
      for (int v = 0; v < 8; ++v) acc[g][v] = (float)xcur[g][v];
    }
#pragma unroll
    for (int g = 0; g < 4; ++g)
#pragma unroll
      for (int kk = 0; kk < 4; ++kk) acc[g] = wmma16(Af[kk], Bf[g][kk], acc[g]);

    __syncthreads();  // all hsh reads done before anyone writes new h

#pragma unroll
    for (int v = 0; v < 8; ++v) {
      float gi = sigm(acc[0][v]);
      float gf = sigm(acc[1][v]);
      float gg = fast_tanh(acc[2][v]);
      float go = sigm(acc[3][v]);
      float cn = gf * cst[v] + gi * gg;
      cst[v] = cn;
      float hv = go * fast_tanh(cn);
      int row = m0 + v + hi8;
      int col = c0 + l15;
      _Float16 h16 = (_Float16)hv;
      hsh[row][col] = h16;
      hout[((size_t)t * 64 + row) * 128 + col] = h16;
    }
#pragma unroll
    for (int g = 0; g < 4; ++g) xcur[g] = xnext[g];
    __syncthreads();  // h visible before next step's reads
  }
}

// ---------------- attention scores: attnS[b][t] = sum_n tanh((h1@weight_W)[r,n]+bias[n])*proj[n] ----------------
__global__ __launch_bounds__(256) void attn_kernel(const _Float16* __restrict__ h1,  // [T*B][128]
                                                   const _Float16* __restrict__ wWt, // [128][128] n-major
                                                   const float* __restrict__ biasv,  // [128]
                                                   const float* __restrict__ proj,   // [128]
                                                   float* __restrict__ attnS) {      // [B][T]
  int gtid = blockIdx.x * blockDim.x + threadIdx.x;
  int lane = gtid & 31;
  int rt = gtid >> 5;
  if (rt >= (TT * BB) / 16) return;
  int r0 = rt * 16;
  int l15 = lane & 15;
  int hi8 = (lane >= 16) ? 8 : 0;
  v16h Af[4];
#pragma unroll
  for (int kk = 0; kk < 4; ++kk) Af[kk] = load_afrag_g(h1, 128, r0 + l15, kk * 32, lane);
  float part[8];
#pragma unroll
  for (int v = 0; v < 8; ++v) part[v] = 0.f;
#pragma unroll
  for (int j = 0; j < 8; ++j) {
    int n = j * 16 + l15;
    v8f acc = {0.f, 0.f, 0.f, 0.f, 0.f, 0.f, 0.f, 0.f};
#pragma unroll
    for (int kk = 0; kk < 4; ++kk)
      acc = wmma16(Af[kk], load_bfrag_g(wWt, 128, n, kk * 32, lane), acc);
    float bn = biasv[n], pj = proj[n];
#pragma unroll
    for (int v = 0; v < 8; ++v) part[v] += fast_tanh(acc[v] + bn) * pj;
  }
#pragma unroll
  for (int mask = 1; mask <= 8; mask <<= 1)
#pragma unroll
    for (int v = 0; v < 8; ++v) part[v] += __shfl_xor(part[v], mask, 32);
  if (l15 == 0) {
#pragma unroll
    for (int v = 0; v < 8; ++v) {
      int r = r0 + v + hi8;                    // r = t*64 + b
      attnS[(size_t)(r & 63) * TT + (r >> 6)] = part[v];
    }
  }
}

// ---------------- softmax over time + context vector per batch ----------------
__global__ __launch_bounds__(256) void ctx_kernel(const float* __restrict__ attnS,
                                                  const _Float16* __restrict__ h1,
                                                  float* __restrict__ ctx) {   // [B][H]
  __shared__ float ebuf[TT];
  __shared__ float red[256];
  int b = blockIdx.x, tid = threadIdx.x;
  const float* arow = attnS + (size_t)b * TT;
  float mx = -3.4e38f;
  for (int t = tid; t < TT; t += 256) mx = fmaxf(mx, arow[t]);
  red[tid] = mx; __syncthreads();
  for (int s = 128; s > 0; s >>= 1) { if (tid < s) red[tid] = fmaxf(red[tid], red[tid + s]); __syncthreads(); }
  mx = red[0]; __syncthreads();
  float sm = 0.f;
  for (int t = tid; t < TT; t += 256) { float e = __expf(arow[t] - mx); ebuf[t] = e; sm += e; }
  red[tid] = sm; __syncthreads();
  for (int s = 128; s > 0; s >>= 1) { if (tid < s) red[tid] += red[tid + s]; __syncthreads(); }
  float inv = 1.f / red[0];
  __syncthreads();
  if (tid < HH) {
    float a0 = 0.f, a1 = 0.f, a2 = 0.f, a3 = 0.f;
    const _Float16* hb = h1 + (size_t)b * 128 + tid;
    for (int t = 0; t < TT; t += 4) {
      a0 += ebuf[t + 0] * (float)hb[(size_t)(t + 0) * 64 * 128];
      a1 += ebuf[t + 1] * (float)hb[(size_t)(t + 1) * 64 * 128];
      a2 += ebuf[t + 2] * (float)hb[(size_t)(t + 2) * 64 * 128];
      a3 += ebuf[t + 3] * (float)hb[(size_t)(t + 3) * 64 * 128];
    }
    ctx[b * HH + tid] = (a0 + a1 + a2 + a3) * inv;
  }
}

// ---------------- classifier head ----------------
__global__ __launch_bounds__(256) void head_kernel(const float* __restrict__ ctx,
                                                   const float* __restrict__ dnn1_w, const float* __restrict__ dnn1_b,
                                                   const float* __restrict__ out_w, const float* __restrict__ out_b,
                                                   float* __restrict__ out) {
  __shared__ float tmp[64][64];
  int tid = threadIdx.x;
  for (int e = tid; e < 64 * 64; e += 256) {
    int bb = e >> 6, j = e & 63;
    float a = dnn1_b[j];
    for (int h = 0; h < HH; ++h) a += ctx[bb * HH + h] * dnn1_w[j * HH + h];
    tmp[bb][j] = a;
  }
  __syncthreads();
  for (int e = tid; e < 64 * NCC; e += 256) {
    int bb = e / NCC, c = e % NCC;
    float a = out_b[c];
    for (int j = 0; j < 64; ++j) a += tmp[bb][j] * out_w[c * 64 + j];
    out[bb * NCC + c] = a;
  }
}

extern "C" void kernel_launch(void* const* d_in, const int* in_sizes, int n_in,
                              void* d_out, int out_size, void* d_ws, size_t ws_size,
                              hipStream_t stream) {
  const float* x        = (const float*)d_in[0];
  const float* W_ih0    = (const float*)d_in[1];
  const float* W_hh0    = (const float*)d_in[2];
  const float* b_ih0    = (const float*)d_in[3];
  const float* b_hh0    = (const float*)d_in[4];
  const float* W_ih1    = (const float*)d_in[5];
  const float* W_hh1    = (const float*)d_in[6];
  const float* b_ih1    = (const float*)d_in[7];
  const float* b_hh1    = (const float*)d_in[8];
  const float* weight_W = (const float*)d_in[9];
  const float* bias     = (const float*)d_in[10];
  const float* w_proj   = (const float*)d_in[11];
  const float* dnn1_w   = (const float*)d_in[12];
  const float* dnn1_b   = (const float*)d_in[13];
  const float* out_w    = (const float*)d_in[14];
  const float* out_b    = (const float*)d_in[15];

  char* ws = (char*)d_ws;
  size_t off = 0;
  auto alloc = [&](size_t bytes) -> void* {
    void* p = ws + off;
    off = (off + bytes + 255) & ~(size_t)255;
    return p;
  };
  _Float16* wih0p  = (_Float16*)alloc((size_t)512 * 64 * 2);
  _Float16* wih1c  = (_Float16*)alloc((size_t)512 * 128 * 2);
  _Float16* whh0c  = (_Float16*)alloc((size_t)512 * 128 * 2);
  _Float16* whh1c  = (_Float16*)alloc((size_t)512 * 128 * 2);
  _Float16* wWt    = (_Float16*)alloc((size_t)128 * 128 * 2);
  float*    bias0c = (float*)alloc(512 * 4);
  float*    bias1c = (float*)alloc(512 * 4);
  _Float16* xpf    = (_Float16*)alloc((size_t)TT * 32 * 4 * 32 * 8 * 2);   // 256 MB
  _Float16* h0     = (_Float16*)alloc((size_t)TT * BB * HH * 2);           // 64 MB
  _Float16* h1     = (_Float16*)alloc((size_t)TT * BB * HH * 2);           // 64 MB
  float*    attnS  = (float*)alloc((size_t)BB * TT * 4);
  float*    ctx    = (float*)alloc((size_t)BB * HH * 4);

  convert_kernel<<<256, 256, 0, stream>>>(W_ih0, W_ih1, W_hh0, W_hh1, weight_W,
                                          b_ih0, b_hh0, b_ih1, b_hh1,
                                          wih0p, wih1c, whh0c, whh1c, wWt, bias0c, bias1c);
  // layer 0
  proj0_kernel<<<(TT * 32 * 32) / 256, 256, 0, stream>>>(x, wih0p, bias0c, xpf);
  lstm_rec_kernel<<<1, 1024, 0, stream>>>(whh0c, xpf, h0);
  // layer 1
  proj1_kernel<<<(TT * 32 * 32) / 256, 256, 0, stream>>>(h0, wih1c, bias1c, xpf);
  lstm_rec_kernel<<<1, 1024, 0, stream>>>(whh1c, xpf, h1);
  // attention + head
  attn_kernel<<<((TT * BB / 16) * 32) / 256, 256, 0, stream>>>(h1, wWt, bias, w_proj, attnS);
  ctx_kernel<<<BB, 256, 0, stream>>>(attnS, h1, ctx);
  head_kernel<<<1, 256, 0, stream>>>(ctx, dnn1_w, dnn1_b, out_w, out_b, (float*)d_out);
}
